// PatchDictionaryLearningTokenized_45801531244953
// MI455X (gfx1250) — matrix operand order repaired and checked by reading, not compiled, not run on hardware
//
#include <hip/hip_runtime.h>
#include <hip/hip_bf16.h>

// ---------------------------------------------------------------------------
// Batched Cholesky OMP for MI455X (gfx1250, wave32, WMMA).
// Phase 1: G = D^T D + eps*I        (1024x1024x1024 GEMM)
// Phase 2: h_bar = X^T D            (8192x1024x1024 GEMM)
// Phase 3: per-batch OMP (argmax / rank-1 Cholesky / residual correlation)
// GEMMs use bf16 split precision (hi+lo) on v_wmma_f32_16x16x32_bf16:
// 3 WMMAs per tile recover ~fp32 accuracy at ~8x the K-depth of f32 WMMA.
// K-loop is software-pipelined: next panel's global_load_b128s are issued
// before the WMMA section so HBM latency overlaps matrix compute.
// ---------------------------------------------------------------------------

typedef __attribute__((ext_vector_type(16))) __bf16 v16bf;
typedef __attribute__((ext_vector_type(8)))  __bf16 v8bf;
typedef __attribute__((ext_vector_type(8)))  float  v8f;

__device__ __forceinline__ float fixf(float v) {
    return __builtin_isfinite(v) ? v : 0.0f;   // nan_to_num(nan/inf -> 0)
}

__device__ __forceinline__ void splitStore(__bf16* hiP, __bf16* loP, float v) {
    v = fixf(v);
    __bf16 hi = (__bf16)v;
    *hiP = hi;
    *loP = (__bf16)(v - (float)hi);
}

// ---------------------------------------------------------------------------
// C[M,N] = sum_k A[k,m] * B[k,n]   (A: [Kd,M] row-major lda, B: [Kd,N] ldb)
// Block: 256 threads = 8 waves (4 M-waves x 2 N-waves); block tile 128x64.
// Wave tile 32x32 = 2x2 WMMA 16x16 tiles. K staged in 32-chunks into LDS as
// bf16 hi/lo planes laid out [col][k] so fragment reads are contiguous b128.
// ---------------------------------------------------------------------------
__global__ __launch_bounds__(256)
void gemm_tn_bf16split(const float* __restrict__ A,
                       const float* __restrict__ B,
                       float* __restrict__ C,
                       int M, int N, int Kd, int lda, int ldb, int ldc,
                       float diagEps)
{
    // [plane hi/lo][col][k], row stride 40 bf16 = 80 B (16B aligned rows)
    __shared__ __align__(16) __bf16 lA[2][128][40];
    __shared__ __align__(16) __bf16 lB[2][64][40];

    const int tid   = threadIdx.x;
    const int lane  = tid & 31;
    const int wave  = tid >> 5;
    const int waveM = wave >> 1;     // 0..3
    const int waveN = wave & 1;      // 0..1
    const int half  = lane >> 4;     // K-half selector (wave32 layout)
    const int l16   = lane & 15;     // M/N within 16x16 tile
    const int cm0   = blockIdx.y * 128;
    const int cn0   = blockIdx.x * 64;

    // Staging coordinates (float4 / global_load_b128 granularity):
    // A panel 32(k) x 128(m): 4 passes of 8 k-rows x 32 float4-cols
    const int acol  = (tid & 31) * 4;
    const int akrow = tid >> 5;          // 0..7
    // B panel 32(k) x 64(n): 2 passes of 16 k-rows x 16 float4-cols
    const int bcol  = (tid & 15) * 4;
    const int bkrow = tid >> 4;          // 0..15

    float4 va[4], vb[2];

#define LOAD_PANELS(K0)                                                        \
    {                                                                          \
        _Pragma("unroll")                                                      \
        for (int i = 0; i < 4; ++i)                                            \
            va[i] = *(const float4*)&A[(size_t)((K0) + akrow + i * 8) * lda +  \
                                       cm0 + acol];                            \
        _Pragma("unroll")                                                      \
        for (int i = 0; i < 2; ++i)                                            \
            vb[i] = *(const float4*)&B[(size_t)((K0) + bkrow + i * 16) * ldb + \
                                       cn0 + bcol];                            \
    }

#define STORE_PANELS()                                                         \
    {                                                                          \
        _Pragma("unroll")                                                      \
        for (int i = 0; i < 4; ++i) {                                          \
            int k = akrow + i * 8;                                             \
            splitStore(&lA[0][acol + 0][k], &lA[1][acol + 0][k], va[i].x);     \
            splitStore(&lA[0][acol + 1][k], &lA[1][acol + 1][k], va[i].y);     \
            splitStore(&lA[0][acol + 2][k], &lA[1][acol + 2][k], va[i].z);     \
            splitStore(&lA[0][acol + 3][k], &lA[1][acol + 3][k], va[i].w);     \
        }                                                                      \
        _Pragma("unroll")                                                      \
        for (int i = 0; i < 2; ++i) {                                          \
            int k = bkrow + i * 16;                                            \
            splitStore(&lB[0][bcol + 0][k], &lB[1][bcol + 0][k], vb[i].x);     \
            splitStore(&lB[0][bcol + 1][k], &lB[1][bcol + 1][k], vb[i].y);     \
            splitStore(&lB[0][bcol + 2][k], &lB[1][bcol + 2][k], vb[i].z);     \
            splitStore(&lB[0][bcol + 3][k], &lB[1][bcol + 3][k], vb[i].w);     \
        }                                                                      \
    }

    v8f acc[2][2] = {};

    LOAD_PANELS(0);                       // prologue: panel 0 -> registers

    for (int k0 = 0; k0 < Kd; k0 += 32) {
        STORE_PANELS();                   // registers -> LDS (fp32 -> hi/lo)
        __syncthreads();

        if (k0 + 32 < Kd)                 // issue next panel loads now;
            LOAD_PANELS(k0 + 32);         // latency hides under WMMAs below

        // ---- load fragments (16-bit A layout: e<8 -> K=half*8+e,
        //      e>=8 -> K=16+half*8+(e-8);  B layout: same as A) ----
        v16bf afrag[2][2], bfrag[2][2];
        #pragma unroll
        for (int mt = 0; mt < 2; ++mt) {
            int m = waveM * 32 + mt * 16 + l16;
            #pragma unroll
            for (int p = 0; p < 2; ++p) {
                v8bf x0 = *(const v8bf*)&lA[p][m][half * 8];
                v8bf x1 = *(const v8bf*)&lA[p][m][16 + half * 8];
                v16bf a;
                #pragma unroll
                for (int e = 0; e < 8; ++e) { a[e] = x0[e]; a[8 + e] = x1[e]; }
                afrag[mt][p] = a;
            }
        }
        #pragma unroll
        for (int nt = 0; nt < 2; ++nt) {
            int n = waveN * 32 + nt * 16 + l16;
            #pragma unroll
            for (int p = 0; p < 2; ++p) {
                v8bf x0 = *(const v8bf*)&lB[p][n][half * 8];
                v8bf x1 = *(const v8bf*)&lB[p][n][16 + half * 8];
                v16bf bb;
                #pragma unroll
                for (int e = 0; e < 8; ++e) { bb[e] = x0[e]; bb[8 + e] = x1[e]; }
                bfrag[nt][p] = bb;
            }
        }

        // ---- 2x2 tiles x 3 split-precision WMMAs ----
        #pragma unroll
        for (int mt = 0; mt < 2; ++mt) {
            #pragma unroll
            for (int nt = 0; nt < 2; ++nt) {
                acc[mt][nt] = __builtin_amdgcn_wmma_f32_16x16x32_bf16(
                    false, afrag[mt][0], false, bfrag[nt][0], (short)0,
                    acc[mt][nt], false, false);
                acc[mt][nt] = __builtin_amdgcn_wmma_f32_16x16x32_bf16(
                    false, afrag[mt][0], false, bfrag[nt][1], (short)0,
                    acc[mt][nt], false, false);
                acc[mt][nt] = __builtin_amdgcn_wmma_f32_16x16x32_bf16(
                    false, afrag[mt][1], false, bfrag[nt][0], (short)0,
                    acc[mt][nt], false, false);
            }
        }
        __syncthreads();                  // LDS consumed; safe to overwrite
    }
#undef LOAD_PANELS
#undef STORE_PANELS

    // ---- epilogue: C/D layout lane<16 -> M=v, lane>=16 -> M=v+8 ----
    #pragma unroll
    for (int mt = 0; mt < 2; ++mt) {
        #pragma unroll
        for (int nt = 0; nt < 2; ++nt) {
            int n = cn0 + waveN * 32 + nt * 16 + l16;
            #pragma unroll
            for (int v = 0; v < 8; ++v) {
                int m = cm0 + waveM * 32 + mt * 16 + v + half * 8;
                float val = acc[mt][nt][v];
                if (diagEps != 0.0f && m == n) val += diagEps;
                if (m < M && n < N) C[(size_t)m * ldc + n] = val;
            }
        }
    }
}

// ---------------------------------------------------------------------------
// OMP: one wave32 per batch element. Each lane owns 8 float4 chunks of the
// 1024-wide correlation row: element (i,c) of lane L is column i*128+L*4+c.
// Cholesky (<=8x8) replicated per-lane, fully unrolled (register-resident).
// ---------------------------------------------------------------------------
__global__ __launch_bounds__(256)
void omp_wave_kernel(const float* __restrict__ G,     // [K][K], eps on diag
                     const float* __restrict__ hbar,  // [B][K]
                     float* __restrict__ outI,        // [B*8] indices (as f32)
                     float* __restrict__ outC,        // [B*8] coefficients
                     int Bt, int K)
{
    const int lane = threadIdx.x & 31;
    const int wave = threadIdx.x >> 5;
    const int b = blockIdx.x * 8 + wave;
    if (b >= Bt) return;

    const float4* hb4 = (const float4*)(hbar + (size_t)b * K);
    float4 hb[8], h[8];
    #pragma unroll
    for (int i = 0; i < 8; ++i) {
        float4 v = hb4[i * 32 + lane];
        v.x = fixf(v.x); v.y = fixf(v.y); v.z = fixf(v.z); v.w = fixf(v.w);
        hb[i] = v; h[i] = v;
    }

    unsigned maskbits = 0;
    int   Isel[8] = {};
    float L[8][8] = {};
    float cf[8]   = {};
    float hsel[8] = {};

    #pragma unroll
    for (int s = 0; s < 8; ++s) {
        // ---- masked argmax |h| (low index wins ties) ----
        float best = -1.0f; int bi = K;
        #pragma unroll
        for (int i = 0; i < 8; ++i) {
            #define OMP_SCORE(VAL, CB)                                        \
                { float a = fabsf(VAL);                                       \
                  if ((maskbits >> (i * 4 + CB)) & 1u) a = -1.0f;             \
                  int n = i * 128 + lane * 4 + CB;                            \
                  if (a > best || (a == best && n < bi)) { best = a; bi = n; } }
            OMP_SCORE(h[i].x, 0) OMP_SCORE(h[i].y, 1)
            OMP_SCORE(h[i].z, 2) OMP_SCORE(h[i].w, 3)
            #undef OMP_SCORE
        }
        #pragma unroll
        for (int off = 16; off >= 1; off >>= 1) {
            float ob = __shfl_xor(best, off, 32);
            int   oi = __shfl_xor(bi,   off, 32);
            if (ob > best || (ob == best && oi < bi)) { best = ob; bi = oi; }
        }
        const int k = bi;
        if (((k >> 2) & 31) == lane)
            maskbits |= 1u << (((k >> 7) << 2) | (k & 3));
        Isel[s] = k;

        const float diag = G[(size_t)k * K + k];          // includes DIAG_EPS
        hsel[s] = hbar[(size_t)b * K + k];

        // ---- Cholesky rank-1 extension ----
        if (s == 0) {
            L[0][0] = sqrtf(fmaxf(diag, 1e-6f));
        } else {
            float w[8] = {};
            float ssum = 0.0f;
            #pragma unroll
            for (int j = 0; j < 8; ++j) if (j < s) {
                float wv = G[(size_t)Isel[j] * K + k];
                #pragma unroll
                for (int p = 0; p < 8; ++p) if (p < j) wv -= L[j][p] * w[p];
                wv /= L[j][j];
                w[j] = wv; L[s][j] = wv; ssum += wv * wv;
            }
            L[s][s] = sqrtf(fmaxf(diag - ssum, 1e-6f));
        }

        // ---- cho_solve: (L L^T) x = hsel  (forward then back) ----
        float y[8] = {}, x[8] = {};
        #pragma unroll
        for (int j = 0; j < 8; ++j) if (j <= s) {
            float v = hsel[j];
            #pragma unroll
            for (int p = 0; p < 8; ++p) if (p < j) v -= L[j][p] * y[p];
            y[j] = v / L[j][j];
        }
        #pragma unroll
        for (int j = 7; j >= 0; --j) if (j <= s) {
            float v = y[j];
            #pragma unroll
            for (int p = 0; p < 8; ++p) if (p > j && p <= s) v -= L[p][j] * x[p];
            float xv = fixf(v / L[j][j]);
            x[j] = xv; cf[j] = xv;
        }

        // ---- h = n2n(h_bar - sum_j cf[j] * G[Isel[j], :]) ----
        #pragma unroll
        for (int i = 0; i < 8; ++i) h[i] = hb[i];
        #pragma unroll
        for (int j = 0; j < 8; ++j) if (j <= s) {
            const float4* g4 = (const float4*)(G + (size_t)Isel[j] * K);
            const float cj = cf[j];
            #pragma unroll
            for (int i = 0; i < 8; ++i) {
                float4 g = g4[i * 32 + lane];
                h[i].x -= cj * g.x; h[i].y -= cj * g.y;
                h[i].z -= cj * g.z; h[i].w -= cj * g.w;
            }
        }
        #pragma unroll
        for (int i = 0; i < 8; ++i) {
            h[i].x = fixf(h[i].x); h[i].y = fixf(h[i].y);
            h[i].z = fixf(h[i].z); h[i].w = fixf(h[i].w);
        }
    }

    if (lane == 0) {
        #pragma unroll
        for (int s = 0; s < 8; ++s) {
            outI[(size_t)b * 8 + s] = (float)Isel[s];
            outC[(size_t)b * 8 + s] = cf[s];
        }
    }
}

// ---------------------------------------------------------------------------
extern "C" void kernel_launch(void* const* d_in, const int* in_sizes, int n_in,
                              void* d_out, int out_size, void* d_ws, size_t ws_size,
                              hipStream_t stream) {
    const float* X = (const float*)d_in[0];   // [dim, B] row-major
    const float* D = (const float*)d_in[1];   // [dim, K] row-major

    const int dim = 1024;
    const int K   = in_sizes[1] / dim;        // 1024
    const int Bt  = in_sizes[0] / dim;        // 8192

    float* G    = (float*)d_ws;               // [K][K]       (4 MB)
    float* hbar = G + (size_t)K * K;          // [Bt][K]      (32 MB)

    dim3 blk(256);

    // G = D^T D + DIAG_EPS * I
    gemm_tn_bf16split<<<dim3(K / 64, K / 128), blk, 0, stream>>>(
        D, D, G, K, K, dim, K, K, K, 1e-4f);

    // h_bar = X^T D   ([Bt][K])
    gemm_tn_bf16split<<<dim3(K / 64, Bt / 128), blk, 0, stream>>>(
        X, D, hbar, Bt, K, dim, Bt, K, K, 0.0f);

    // OMP sweep: outputs = (I as float)[Bt*8] ++ coeffs[Bt*8]
    float* outI = (float*)d_out;
    float* outC = outI + (size_t)Bt * 8;
    omp_wave_kernel<<<dim3(Bt / 8), blk, 0, stream>>>(G, hbar, outI, outC, Bt, K);
}
